// S5SSM_42743514529989
// MI455X (gfx1250) — compile-verified, matrix operand order checked by reading, not run on hardware
//
#include <hip/hip_runtime.h>
#include <hip/hip_bf16.h>

// ---------------------------------------------------------------------------
// S5 SSM forward for MI455X (gfx1250, wave32).
//   L=131072, H=256, P=64 complex states (128 real state columns).
//   K1: precompute Lambda_bar, B_bar^T (bf16), C weights (bf16, x2 folded in)
//   K2: per 256-step chunk: async-double-buffered u tiles (HBM->LDS via
//       global_load_async_to_lds_b128), Bu = u @ Bbar^T via
//       v_wmma_f32_16x16x32_bf16, then in-chunk scan (bf16 state dump)
//   K3: serial scan over 512 chunk summaries (closed-form Lambda_bar^CH)
//   K4: async-load bf16 state tiles to LDS, closed-form fixup
//       Lambda_bar^(t+1)*x_in (exp/sincos), output GEMM (bf16 WMMA) + u*D
// Workspace: ~34.5 MB (x_local bf16 dominates).
// ---------------------------------------------------------------------------

typedef __bf16 bf16;
typedef __attribute__((ext_vector_type(16))) __bf16 v16bf;
typedef __attribute__((ext_vector_type(8)))  float  v8f;

#define LSEQ 131072
#define HD   256
#define PS   64     // complex states
#define NW   128    // real state columns (re | im)
#define CH   256    // timesteps per chunk
#define GCH  (LSEQ / CH)   // 512 chunks

// Async HBM->LDS copy, 16B per lane, ASYNCcnt-tracked (ISA 15.18.3 op 98).
// GVS mode: 64-bit SGPR base + 32-bit per-lane VGPR byte offset.
__device__ __forceinline__ void async_copy_b128(unsigned int lds_off,
                                                const void* gbase, int voff) {
  asm volatile("global_load_async_to_lds_b128 %0, %1, %2"
               :: "v"(lds_off), "v"(voff), "s"(gbase) : "memory");
}
__device__ __forceinline__ void wait_async_le4() {
  asm volatile("s_wait_asynccnt 0x4" ::: "memory");
}
__device__ __forceinline__ void wait_async_le1() {
  asm volatile("s_wait_asynccnt 0x1" ::: "memory");
}
__device__ __forceinline__ void wait_async_0() {
  asm volatile("s_wait_asynccnt 0x0" ::: "memory");
}
__device__ __forceinline__ unsigned int lds_addr(const void* p) {
  return (unsigned int)(unsigned long long)p;   // LDS aperture: addr[31:0]
}

// 16x32 bf16 WMMA fragment (A or B role) from row-major storage.
// ISA 7.12.2: lanes 0-15 hold K runs {0..7,16..23}; lanes 16-31 {8..15,24..31}.
__device__ __forceinline__ v16bf load_frag(const bf16* m, int stride, int rowbase, int kb) {
  int l  = threadIdx.x & 31;
  int r  = l & 15;
  int kg = l >> 4;
  const bf16* p = m + (size_t)(rowbase + r) * stride + kb + kg * 8;
  union { uint4 q[2]; v16bf v; } un;
  un.q[0] = *(const uint4*)(p);
  un.q[1] = *(const uint4*)(p + 16);
  return un.v;
}

// ---------------------------------------------------------------------------
// Kernel 1: setup.  Wt[n][h] = Bbar_re|im (n-major, bf16);  Cwt[h][j] =
// {2*C_re | -2*C_im} (h-major, bf16);  ldel = Lambda*delta; abar = exp(ldel);
// afull = exp(ldel*CH).
// ---------------------------------------------------------------------------
__global__ __launch_bounds__(256) void s5_setup(
    const float* __restrict__ Lre, const float* __restrict__ Lim,
    const float* __restrict__ Bre, const float* __restrict__ Bim,
    const float* __restrict__ Cre, const float* __restrict__ Cim,
    const float* __restrict__ deltas,
    bf16* __restrict__ Wt, bf16* __restrict__ Cwt,
    float* __restrict__ ldel, float* __restrict__ abar, float* __restrict__ afull)
{
  int tid = threadIdx.x;
  if (tid < PS) {
    int p = tid;
    float lr = Lre[p] * deltas[p], li = Lim[p] * deltas[p];
    ldel[p] = lr; ldel[PS + p] = li;
    float er = __expf(lr); float s, c; __sincosf(li, &s, &c);
    abar[p] = er * c; abar[PS + p] = er * s;
    float e2 = __expf(lr * (float)CH); float s2, c2; __sincosf(li * (float)CH, &s2, &c2);
    afull[p] = e2 * c2; afull[PS + p] = e2 * s2;
  }
  for (int i = tid; i < NW * HD; i += blockDim.x) {
    int n = i / HD, h = i - n * HD;
    int p = n & 63;
    float lr0 = Lre[p], li0 = Lim[p];
    float lr = lr0 * deltas[p], li = li0 * deltas[p];
    float er = __expf(lr); float s, c; __sincosf(li, &s, &c);
    float ar = er * c, ai = er * s;
    float den = lr0 * lr0 + li0 * li0;
    float sr = ((ar - 1.f) * lr0 + ai * li0) / den;   // (a-1)/Lambda
    float si = (ai * lr0 - (ar - 1.f) * li0) / den;
    float br = Bre[p * HD + h], bi = Bim[p * HD + h];
    float w = (n < 64) ? (sr * br - si * bi) : (sr * bi + si * br);
    Wt[i] = (bf16)w;
  }
  for (int i = tid; i < HD * NW; i += blockDim.x) {
    int h = i >> 7, j = i & 127;
    int p = j & 63;
    float v = (j < 64) ? (2.f * Cre[h * PS + p]) : (-2.f * Cim[h * PS + p]);
    Cwt[i] = (bf16)v;
  }
}

// ---------------------------------------------------------------------------
// Kernel 2: per-chunk  Bu = u @ Bbar^T  (WMMA bf16)  +  in-chunk scan.
// Dynamic LDS: WtL 64K | uF32[2] 32K | uA 8K | BuL 128K | dL 256B  (~232KB)
// u tiles stream HBM->LDS through the async path, double buffered.
// ---------------------------------------------------------------------------
__global__ __launch_bounds__(256) void s5_chunk_gemm_scan(
    const float* __restrict__ u, const unsigned char* __restrict__ dflags,
    const bf16* __restrict__ Wt, const float* __restrict__ abar,
    bf16* __restrict__ xloc, float* __restrict__ bsum, int* __restrict__ rfirst)
{
  extern __shared__ char smem[];
  bf16*  WtL = (bf16*)smem;                              // 128*256 bf16 = 64KB
  float* uF0 = (float*)(smem + 65536);                   // 16*256 f32 = 16KB
  float* uF1 = (float*)(smem + 65536 + 16384);           // 16KB
  bf16*  uA  = (bf16*)(smem + 65536 + 32768);            // 16*256 bf16 = 8KB
  float* BuL = (float*)(smem + 65536 + 32768 + 8192);    // 256*128 f32 = 128KB
  unsigned char* dL = (unsigned char*)(smem + 65536 + 32768 + 8192 + 131072);

  int tid = threadIdx.x;
  int c   = blockIdx.x;
  long base_t = (long)c * CH;

  { // Bbar^T weights into LDS (b128, coalesced)
    const uint4* src = (const uint4*)Wt;
    uint4* dst = (uint4*)WtL;
    for (int i = tid; i < (NW * HD * 2) / 16; i += 256) dst[i] = src[i];
  }
  dL[tid] = dflags[base_t + tid];   // CH == blockDim.x == 256

  int wave = tid >> 5;
  int nb   = wave * 16;             // each wave owns one 16-wide N tile of Bu
  const int NT = CH / 16;

  // prologue: async-issue tile 0 (4 instr/wave: 16KB tile, 16B/lane)
  {
    const float* g = u + base_t * HD;
    unsigned int l0 = lds_addr(uF0);
    for (int s = tid; s < 1024; s += 256)
      async_copy_b128(l0 + s * 16, g, s * 16);
  }

  for (int tt = 0; tt < NT; ++tt) {
    if (tt + 1 < NT) {              // issue tile tt+1, then wait for tile tt
      const float* g = u + (base_t + (long)(tt + 1) * 16) * HD;
      unsigned int l1 = lds_addr(((tt + 1) & 1) ? uF1 : uF0);
      for (int s = tid; s < 1024; s += 256)
        async_copy_b128(l1 + s * 16, g, s * 16);
      wait_async_le4();             // ASYNCcnt in-order: older 4 done
    } else {
      wait_async_0();
    }
    __syncthreads();                // tile visible; prev WMMA done with uA
    const float* src = (tt & 1) ? uF1 : uF0;
    for (int i = tid; i < 16 * HD; i += 256) uA[i] = (bf16)src[i];
    __syncthreads();

    v8f acc = {};
    #pragma unroll
    for (int kb = 0; kb < HD; kb += 32) {
      v16bf a = load_frag(uA,  HD, 0,  kb);
      v16bf b = load_frag(WtL, HD, nb, kb);
      acc = __builtin_amdgcn_wmma_f32_16x16x32_bf16(false, a, false, b, (short)0, acc, false, false);
    }
    int l = tid & 31, n = l & 15, mh = (l >> 4) * 8;
    float* out = BuL + (tt * 16 + mh) * NW + nb + n;
    #pragma unroll
    for (int r = 0; r < 8; ++r) out[r * NW] = acc[r];
  }
  __syncthreads();

  // in-chunk zero-input scan: lane = complex state, waves 0-1 only
  if (tid < PS) {
    int p = tid;
    float ar = abar[p], ai = abar[PS + p];
    float xr = 0.f, xi = 0.f;
    int rf = CH;
    for (int t = 0; t < CH; ++t) {
      float br = BuL[t * NW + p], bi = BuL[t * NW + PS + p];
      if (dL[t]) {
        xr = br; xi = bi;
        if (rf == CH) rf = t;
      } else {
        float nr = ar * xr - ai * xi + br;
        float ni = ar * xi + ai * xr + bi;
        xr = nr; xi = ni;
      }
      long row = (base_t + t) * (long)NW;   // coalesced b16 stores
      xloc[row + p] = (bf16)xr; xloc[row + PS + p] = (bf16)xi;
    }
    bsum[c * NW + p] = xr; bsum[c * NW + PS + p] = xi;
    if (p == 0) rfirst[c] = rf;
  }
}

// ---------------------------------------------------------------------------
// Kernel 3: serial scan over chunk summaries (512 steps, 64 lanes).
// ---------------------------------------------------------------------------
__global__ void s5_chunkscan(
    const float* __restrict__ bsum, const int* __restrict__ rfirst,
    const float* __restrict__ afull,
    const float* __restrict__ x0re, const float* __restrict__ x0im,
    float* __restrict__ xin)
{
  int p = threadIdx.x;
  if (p >= PS) return;
  float xr = x0re[p], xi = x0im[p];
  float ar = afull[p], ai = afull[PS + p];
  for (int c = 0; c < GCH; ++c) {
    xin[c * NW + p] = xr; xin[c * NW + PS + p] = xi;
    float br = bsum[c * NW + p], bi = bsum[c * NW + PS + p];
    if (rfirst[c] < CH) { xr = br; xi = bi; }
    else {
      float nr = ar * xr - ai * xi + br;
      float ni = ar * xi + ai * xr + bi;
      xr = nr; xi = ni;
    }
  }
}

// ---------------------------------------------------------------------------
// Kernel 4: async-load bf16 state tiles, closed-form fixup, output GEMM
// (bf16 WMMA) + u*D epilogue.  Dynamic LDS: CwL 64K | bufX[2] 8K | xsA 4K.
// ---------------------------------------------------------------------------
__global__ __launch_bounds__(256) void s5_out_gemm(
    const bf16* __restrict__ xloc, const float* __restrict__ xin,
    const int* __restrict__ rfirst, const float* __restrict__ ldel,
    const bf16* __restrict__ Cwt, const float* __restrict__ u,
    const float* __restrict__ D, float* __restrict__ y)
{
  extern __shared__ char smem[];
  bf16* CwL  = (bf16*)smem;                      // 256*128 bf16 = 64KB
  bf16* bufX0 = (bf16*)(smem + 65536);           // 16*128 bf16 = 4KB
  bf16* bufX1 = (bf16*)(smem + 65536 + 4096);    // 4KB
  bf16* xsA  = (bf16*)(smem + 65536 + 8192);     // 4KB

  int tid = threadIdx.x;
  int c   = blockIdx.x;
  long base_t = (long)c * CH;

  {
    const uint4* src = (const uint4*)Cwt;
    uint4* dst = (uint4*)CwL;
    for (int i = tid; i < (HD * NW * 2) / 16; i += 256) dst[i] = src[i];
  }
  int rf   = rfirst[c];
  int wave = tid >> 5;
  const int NT = CH / 16;

  // prologue: async-issue state tile 0 (1 instr/wave: 4KB tile, 16B/lane)
  async_copy_b128(lds_addr(bufX0) + tid * 16, xloc + base_t * NW, tid * 16);

  for (int tt = 0; tt < NT; ++tt) {
    if (tt + 1 < NT) {
      const bf16* g = xloc + (base_t + (long)(tt + 1) * 16) * NW;
      bf16* buf = ((tt + 1) & 1) ? bufX1 : bufX0;
      async_copy_b128(lds_addr(buf) + tid * 16, g, tid * 16);
      wait_async_le1();
    } else {
      wait_async_0();
    }
    __syncthreads();
    // fixup pass: x += Lambda_bar^(tl+1) * x_in (closed form), re-emit bf16
    const bf16* buf = (tt & 1) ? bufX1 : bufX0;
    for (int i = tid; i < 16 * NW; i += 256) {
      int tl = tt * 16 + (i >> 7);
      int j  = i & 127;
      float v = (float)buf[i];
      if (tl < rf) {
        int p = j & 63;
        float k = (float)(tl + 1);
        float er = __expf(ldel[p] * k);
        float s, co; __sincosf(ldel[PS + p] * k, &s, &co);
        float pr = er * co, pi = er * s;
        float xr = xin[c * NW + p], xi2 = xin[c * NW + PS + p];
        v += (j < 64) ? (pr * xr - pi * xi2) : (pr * xi2 + pi * xr);
      }
      xsA[i] = (bf16)v;
    }
    __syncthreads();

    #pragma unroll
    for (int half = 0; half < 2; ++half) {  // 16 N-tiles over H=256, 8 waves
      int hb = (wave + half * 8) * 16;
      v8f acc = {};
      #pragma unroll
      for (int kb = 0; kb < NW; kb += 32) {
        v16bf a = load_frag(xsA, NW, 0,  kb);
        v16bf b = load_frag(CwL, NW, hb, kb);
        acc = __builtin_amdgcn_wmma_f32_16x16x32_bf16(false, a, false, b, (short)0, acc, false, false);
      }
      int l = tid & 31, n = l & 15, mh = (l >> 4) * 8;
      long trow = base_t + tt * 16 + mh;
      float dv = D[hb + n];
      #pragma unroll
      for (int r = 0; r < 8; ++r) {
        long idx = (trow + r) * (long)HD + hb + n;
        y[idx] = acc[r] + u[idx] * dv;   // 2x folded into Cw
      }
    }
  }
}

// ---------------------------------------------------------------------------
extern "C" void kernel_launch(void* const* d_in, const int* in_sizes, int n_in,
                              void* d_out, int out_size, void* d_ws, size_t ws_size,
                              hipStream_t stream) {
  const float* u      = (const float*)d_in[0];
  const float* Lre    = (const float*)d_in[1];
  const float* Lim    = (const float*)d_in[2];
  const float* Bre    = (const float*)d_in[3];
  const float* Bim    = (const float*)d_in[4];
  const float* Cre    = (const float*)d_in[5];
  const float* Cim    = (const float*)d_in[6];
  const float* Dv     = (const float*)d_in[7];
  const float* deltas = (const float*)d_in[8];
  const float* x0re   = (const float*)d_in[9];
  const float* x0im   = (const float*)d_in[10];
  const unsigned char* dfl = (const unsigned char*)d_in[11];  // jnp bool = 1B
  float* y = (float*)d_out;

  // workspace carve (~34.5 MB; bf16 x_local dominates)
  char* ws = (char*)d_ws;
  size_t o = 0;
  bf16* Wt    = (bf16*)(ws + o); o += (size_t)NW * HD * 2;        // 64KB
  bf16* Cwt   = (bf16*)(ws + o); o += (size_t)HD * NW * 2;        // 64KB
  float* ldel = (float*)(ws + o); o += NW * 4;
  float* abar = (float*)(ws + o); o += NW * 4;
  float* afull= (float*)(ws + o); o += NW * 4;
  o = (o + 255) & ~(size_t)255;
  float* bsum = (float*)(ws + o); o += (size_t)GCH * NW * 4;      // 256KB
  int* rfirst = (int*)(ws + o);   o += (size_t)GCH * 4;
  o = (o + 255) & ~(size_t)255;
  float* xin  = (float*)(ws + o); o += (size_t)GCH * NW * 4;      // 256KB
  o = (o + 255) & ~(size_t)255;
  bf16* xloc  = (bf16*)(ws + o);  o += (size_t)LSEQ * NW * 2;     // 32MB

  s5_setup<<<1, 256, 0, stream>>>(Lre, Lim, Bre, Bim, Cre, Cim, deltas,
                                  Wt, Cwt, ldel, abar, afull);

  size_t lds2 = 65536 + 32768 + 8192 + (size_t)CH * NW * 4 + 256; // ~232KB of 320KB
  s5_chunk_gemm_scan<<<GCH, 256, lds2, stream>>>(u, dfl, Wt, abar, xloc, bsum, rfirst);

  s5_chunkscan<<<1, 64, 0, stream>>>(bsum, rfirst, afull, x0re, x0im, xin);

  size_t lds4 = 65536 + 4096 + 4096 + 4096;
  s5_out_gemm<<<GCH, 256, lds4, stream>>>(xloc, xin, rfirst, ldel, Cwt, u, Dv, y);
}
